// WeaklySelector_69612829933734
// MI455X (gfx1250) — compile-verified, matrix operand order checked by reading, not run on hardware
//
#include <hip/hip_runtime.h>

#define BB 32
#define SS 4096
#define CC 384
#define KCLS 200
#define NSEL 256
#define NTILES 13      // ceil(200/16)
#define KCHUNKS 12     // 384/32
#define WPACK_TILES (NTILES * KCHUNKS)   // 156 B-operand tiles
#define JSPLIT 8
#define JCH (SS / JSPLIT)                // 512 j-elements per rank block

typedef __attribute__((ext_vector_type(16))) __bf16 v16bf;
typedef __attribute__((ext_vector_type(8)))  float  v8f;

static __device__ __forceinline__ __bf16 f2bf(float f) { return (__bf16)f; }

// ---------------------------------------------------------------------------
// Kernel 0: pre-pack W (384x200 f32) into WMMA B-operand layout, bf16.
// For tile (nt, kk), lane l holds 16 bf16 = 32 contiguous bytes:
//   n = l&15, g = l>>4 ; elems 0..7 -> K = kk*32 + g*8 + e
//                        elems 8..15 -> K = kk*32 + 16 + g*8 + e
// Classes >= 200 are zero-padded so the GEMM loop has no branches.
// ---------------------------------------------------------------------------
__global__ void __launch_bounds__(256)
pack_w_kernel(const float* __restrict__ W, __bf16* __restrict__ wpack) {
    const int id = blockIdx.x * blockDim.x + threadIdx.x;  // [0, WPACK_TILES*32)
    if (id >= WPACK_TILES * 32) return;
    const int lane = id & 31;
    const int kk   = (id >> 5) % KCHUNKS;
    const int nt   = (id >> 5) / KCHUNKS;
    const int n    = lane & 15;
    const int g    = lane >> 4;
    const int kcls = nt * 16 + n;

    v16bf bm = {};
    if (kcls < KCLS) {
        const float* wc0 = W + (size_t)(kk * 32 + g * 8) * KCLS + kcls;
        const float* wc1 = W + (size_t)(kk * 32 + 16 + g * 8) * KCLS + kcls;
#pragma unroll
        for (int e = 0; e < 8; ++e) bm[e]     = f2bf(wc0[(size_t)e * KCLS]);
#pragma unroll
        for (int e = 0; e < 8; ++e) bm[8 + e] = f2bf(wc1[(size_t)e * KCLS]);
    }
    *(v16bf*)(wpack + (size_t)id * 16) = bm;
}

// ---------------------------------------------------------------------------
// Kernel 1: logits = x @ W + b via WMMA bf16. One wave per 16-row s-tile.
// A register-resident across all K. Per N-tile: load ALL 12 B tiles into a
// register array (one 24-load clause, one wait), then fire 12 accumulating
// WMMAs back-to-back — no per-WMMA load waits.
// ---------------------------------------------------------------------------
__global__ void __launch_bounds__(256)
gemm_logits_kernel(const float* __restrict__ x, const __bf16* __restrict__ wpack,
                   const float* __restrict__ bias, float* __restrict__ logits) {
    const int wavesPerBlk = blockDim.x >> 5;
    const int waveId = threadIdx.x >> 5;
    const int lane   = threadIdx.x & 31;
    const int tile   = blockIdx.x * wavesPerBlk + waveId;   // wave-uniform
    if (tile >= (BB * SS) / 16) return;                     // uniform: EXEC stays all-1s

    const int mrow = lane & 15;   // A row / B column / D column selector
    const int grp  = lane >> 4;   // K-half selector

    // ---- Load A tile (16 x 384 fp32) once, convert to bf16 WMMA layout ----
    v16bf areg[KCHUNKS];
    const float* xrow = x + (size_t)(tile * 16 + mrow) * CC;
#pragma unroll
    for (int kk = 0; kk < KCHUNKS; ++kk) {
        const float* p0 = xrow + kk * 32 + grp * 8;        // 32B aligned
        const float* p1 = xrow + kk * 32 + 16 + grp * 8;
        const float4 f0 = *(const float4*)(p0);
        const float4 f1 = *(const float4*)(p0 + 4);
        const float4 f2 = *(const float4*)(p1);
        const float4 f3 = *(const float4*)(p1 + 4);
        v16bf a;
        a[0]  = f2bf(f0.x); a[1]  = f2bf(f0.y); a[2]  = f2bf(f0.z); a[3]  = f2bf(f0.w);
        a[4]  = f2bf(f1.x); a[5]  = f2bf(f1.y); a[6]  = f2bf(f1.z); a[7]  = f2bf(f1.w);
        a[8]  = f2bf(f2.x); a[9]  = f2bf(f2.y); a[10] = f2bf(f2.z); a[11] = f2bf(f2.w);
        a[12] = f2bf(f3.x); a[13] = f2bf(f3.y); a[14] = f2bf(f3.z); a[15] = f2bf(f3.w);
        areg[kk] = a;
    }

    const v16bf* wp = (const v16bf*)wpack;   // 32B per lane per (nt,kk) tile

    // ---- Loop over the 13 N-tiles of the 200 classes ----
    for (int nt = 0; nt < NTILES; ++nt) {
        const v16bf* bp = wp + (size_t)(nt * KCHUNKS) * 32 + lane;

        // Stage all 12 B tiles: one deep load clause, single wait.
        v16bf breg[KCHUNKS];
#pragma unroll
        for (int kk = 0; kk < KCHUNKS; ++kk) breg[kk] = bp[(size_t)kk * 32];

        // 12 accumulating WMMAs back-to-back (D->C chaining, no A/B hazard).
        v8f acc = {};
#pragma unroll
        for (int kk = 0; kk < KCHUNKS; ++kk)
            acc = __builtin_amdgcn_wmma_f32_16x16x32_bf16(
                    false, areg[kk], false, breg[kk], (short)0, acc, false, false);

        // D layout: lane L holds column n=L&15, rows m = v + 8*(L>>4), v=0..7
        const int kcls = nt * 16 + mrow;
        if (kcls < KCLS) {
            const float bv = bias[kcls];
#pragma unroll
            for (int v = 0; v < 8; ++v) {
                const int row = tile * 16 + v + 8 * grp;
                logits[(size_t)row * KCLS + kcls] = acc[v] + bv;
            }
        }
    }
}

// ---------------------------------------------------------------------------
// Kernel 2: conf[p] = max softmax prob = 1 / sum_k exp(logit - max). Wave/pos.
// ---------------------------------------------------------------------------
__global__ void __launch_bounds__(256)
conf_kernel(const float* __restrict__ logits, float* __restrict__ conf) {
    const int waveId = threadIdx.x >> 5;
    const int lane   = threadIdx.x & 31;
    const int p = blockIdx.x * (blockDim.x >> 5) + waveId;
    if (p >= BB * SS) return;
    const float* lp = logits + (size_t)p * KCLS;

    float mx = -3.402823466e38f;
    for (int k = lane; k < KCLS; k += 32) mx = fmaxf(mx, lp[k]);
#pragma unroll
    for (int off = 16; off > 0; off >>= 1) mx = fmaxf(mx, __shfl_xor(mx, off, 32));

    float s = 0.0f;
    for (int k = lane; k < KCLS; k += 32) s += __expf(lp[k] - mx);
#pragma unroll
    for (int off = 16; off > 0; off >>= 1) s += __shfl_xor(s, off, 32);

    if (lane == 0) conf[p] = 1.0f / s;
}

// ---------------------------------------------------------------------------
// Kernel 3a: partial rank counting, split over j-chunks for parallelism.
// Block (b, jc): for every i in [0,4096), count j in the jc-th 512-chunk with
//   conf[j] > conf[i] || (conf[j]==conf[i] && j < i)   (stable descending)
// ---------------------------------------------------------------------------
__global__ void __launch_bounds__(1024)
rank_part_kernel(const float* __restrict__ conf, int* __restrict__ rpart) {
    __shared__ float cj_s[JCH];
    const int b  = blockIdx.x / JSPLIT;
    const int jc = blockIdx.x % JSPLIT;
    const int jbase = jc * JCH;
    const float* cp = conf + (size_t)b * SS;

    for (int t = threadIdx.x; t < JCH; t += blockDim.x) cj_s[t] = cp[jbase + t];
    __syncthreads();

    for (int i = threadIdx.x; i < SS; i += blockDim.x) {
        const float ci = cp[i];
        int r = 0;
        for (int j = 0; j < JCH; ++j) {
            const float cj = cj_s[j];
            r += (cj > ci) || (cj == ci && (jbase + j) < i);
        }
        rpart[(size_t)jc * (BB * SS) + (size_t)b * SS + i] = r;
    }
}

// ---------------------------------------------------------------------------
// Kernel 3b: reduce partial ranks and scatter: order[b, rank] = i
// ---------------------------------------------------------------------------
__global__ void __launch_bounds__(256)
scatter_kernel(const int* __restrict__ rpart, int* __restrict__ order) {
    const int idx = blockIdx.x * blockDim.x + threadIdx.x;   // b*SS + i
    if (idx >= BB * SS) return;
    int r = 0;
#pragma unroll
    for (int jc = 0; jc < JSPLIT; ++jc) r += rpart[(size_t)jc * (BB * SS) + idx];
    const int b = idx >> 12;          // / SS
    const int i = idx & (SS - 1);     // % SS
    order[(size_t)b * SS + r] = i;
}

// ---------------------------------------------------------------------------
// Kernel 4: gathers. out = [ sel(B,256,384) | preds_1(B,256,200) | preds_0(B,3840,200) ]
// ---------------------------------------------------------------------------
__global__ void __launch_bounds__(256)
gather_kernel(const float* __restrict__ x, const float* __restrict__ logits,
              const int* __restrict__ order, float* __restrict__ out) {
    const size_t n1 = (size_t)BB * NSEL * CC;
    const size_t n2 = (size_t)BB * NSEL * KCLS;
    const size_t n3 = (size_t)BB * (SS - NSEL) * KCLS;
    size_t idx = (size_t)blockIdx.x * blockDim.x + threadIdx.x;
    if (idx >= n1 + n2 + n3) return;

    if (idx < n1) {                       // sel = x rows at top indices
        const int c = (int)(idx % CC);
        const size_t r = idx / CC;
        const int i = (int)(r % NSEL);
        const int b = (int)(r / NSEL);
        const int s = order[(size_t)b * SS + i];
        out[idx] = x[((size_t)b * SS + s) * CC + c];
    } else if (idx < n1 + n2) {           // preds_1 = logits at top indices
        const size_t t = idx - n1;
        const int k = (int)(t % KCLS);
        const size_t r = t / KCLS;
        const int i = (int)(r % NSEL);
        const int b = (int)(r / NSEL);
        const int s = order[(size_t)b * SS + i];
        out[idx] = logits[((size_t)b * SS + s) * KCLS + k];
    } else {                              // preds_0 = logits at rest indices
        const size_t t = idx - n1 - n2;
        const int k = (int)(t % KCLS);
        const size_t r = t / KCLS;
        const int j = (int)(r % (SS - NSEL));
        const int b = (int)(r / (SS - NSEL));
        const int s = order[(size_t)b * SS + NSEL + j];
        out[idx] = logits[((size_t)b * SS + s) * KCLS + k];
    }
}

// ---------------------------------------------------------------------------
extern "C" void kernel_launch(void* const* d_in, const int* in_sizes, int n_in,
                              void* d_out, int out_size, void* d_ws, size_t ws_size,
                              hipStream_t stream) {
    (void)in_sizes; (void)n_in; (void)out_size; (void)ws_size;
    const float* x    = (const float*)d_in[0];
    const float* W    = (const float*)d_in[1];
    const float* bias = (const float*)d_in[2];
    float* out = (float*)d_out;

    // Workspace layout:
    //   logits (B*S*200 f32) | conf (B*S f32) | order (B*S i32)
    //   | wpack (13*12*32*16 bf16, 32B-aligned) | rpart (JSPLIT*B*S i32)
    float*  logits = (float*)d_ws;
    float*  conf   = logits + (size_t)BB * SS * KCLS;
    int*    order  = (int*)(conf + (size_t)BB * SS);
    __bf16* wpack  = (__bf16*)(order + (size_t)BB * SS);
    int*    rpart  = (int*)(wpack + (size_t)WPACK_TILES * 32 * 16);

    // 0) Pack W into WMMA B-operand layout (bf16, zero-padded to 13 tiles)
    const int packThreads = WPACK_TILES * 32;          // 4992
    pack_w_kernel<<<(packThreads + 255) / 256, 256, 0, stream>>>(W, wpack);

    // 1) GEMM+bias via WMMA bf16: 8192 s-tiles, 8 waves (256 thr) per block
    const int stiles = (BB * SS) / 16;                 // 8192
    gemm_logits_kernel<<<stiles / 8, 256, 0, stream>>>(x, wpack, bias, logits);

    // 2) Confidence per position: one wave each
    conf_kernel<<<(BB * SS) / 8, 256, 0, stream>>>(logits, conf);

    // 3) Per-batch stable descending argsort: partial ranks then scatter
    rank_part_kernel<<<BB * JSPLIT, 1024, 0, stream>>>(conf, rpart);
    scatter_kernel<<<(BB * SS + 255) / 256, 256, 0, stream>>>(rpart, order);

    // 4) Gather outputs
    const size_t total = (size_t)BB * NSEL * CC
                       + (size_t)BB * NSEL * KCLS
                       + (size_t)BB * (SS - NSEL) * KCLS;
    gather_kernel<<<(unsigned)((total + 255) / 256), 256, 0, stream>>>(x, logits, order, out);
}
